// DepthwiseXCorr_39170101740159
// MI455X (gfx1250) — matrix operand order repaired and checked by reading, not compiled, not run on hardware
//
#include <hip/hip_runtime.h>
#include <hip/hip_bf16.h>

// ---------------------------------------------------------------------------
// SiamRPN depthwise-xcorr head for MI455X (gfx1250, wave32, WMMA).
//
//   1) bn_prep        : fold BN into per-channel scale/shift   (x2)
//   2) pack_weights   : fp32 OIHW -> bf16 WMMA-A fragment order (x2)
//   3) conv_bn_relu   : implicit-GEMM conv3x3 via v_wmma_f32_16x16x32_bf16
//   4) xcorr          : per-(b,c) 29x29 (x) 5x5 -> 25x25, VALU from LDS
//
// GEMM mapping per workgroup (256 thr = 8 waves):
//   M = cout (16 tiles of 16), N = NSP*16 output spatial positions,
//   K = cin*9taps (chunks of 32 = one tap x 32 cins).
//   Each wave: 2 cout tiles x NSP spatial tiles -> 2*NSP accumulators; each
//   A (weight) fragment is reused across NSP WMMAs to cut L2 weight traffic.
//   B (activations) staged fp32->bf16 into LDS as [row][col][cin32]; each
//   lane's 16 bf16 are contiguous & 32B aligned -> 2x ds_load_b128.
// ---------------------------------------------------------------------------

typedef __attribute__((ext_vector_type(16))) __bf16 v16bf;
typedef __attribute__((ext_vector_type(8)))  float  v8f;

#define NCH   256
#define NB    128
#define BN_EPS 1e-5f

// ---------------- BN fold: scale = g*rsqrt(v+eps), shift = b - m*scale -----
__global__ void bn_prep_kernel(const float* __restrict__ gamma,
                               const float* __restrict__ beta,
                               const float* __restrict__ mean,
                               const float* __restrict__ var,
                               float* __restrict__ scale,
                               float* __restrict__ shift) {
  int i = blockIdx.x * blockDim.x + threadIdx.x;
  if (i < NCH) {
    float s = gamma[i] * rsqrtf(var[i] + BN_EPS);
    scale[i] = s;
    shift[i] = beta[i] - mean[i] * s;
  }
}

// ---------------- Weight packing into WMMA A-fragment order ----------------
// K-chunk q = tap*8 + c32  (tap = kh*3+kw, c32 = 32-cin group), 72 chunks.
// A element (lane l, elem e) -> K index, contiguous (same order as B in LDS):
//   m = l&15 (cout in tile), khalf = l>>4,
//   v = e>>1, p = e&1, Kidx = khalf*16 + v*2 + p (= 16*khalf + e)
// Packed linear index: (((q*16 + j)*32 + l)*16 + e)
__global__ void pack_weights_kernel(const float* __restrict__ w,  // [256,256,3,3]
                                    __bf16* __restrict__ apack) { // 72*16*32*16
  int idx = blockIdx.x * blockDim.x + threadIdx.x;
  if (idx >= 72 * 16 * 32 * 16) return;
  int e = idx & 15;
  int l = (idx >> 4) & 31;
  int j = (idx >> 9) & 15;
  int q = idx >> 13;
  int tap = q >> 3;
  int c32 = q & 7;
  int cout  = j * 16 + (l & 15);
  int khalf = l >> 4;
  int v = e >> 1, p = e & 1;
  int kidx = khalf * 16 + v * 2 + p;
  int cin = c32 * 32 + kidx;
  int kh = tap / 3, kw = tap % 3;
  float val = w[(((size_t)cout * NCH + cin) * 3 + kh) * 3 + kw];
  apack[idx] = (__bf16)val;
}

// ---------------- Implicit-GEMM conv3x3 + BN + ReLU (WMMA bf16) ------------
// HIN x HIN -> HOUT x HOUT (VALID). RMAX = max LDS patch rows.
// NSP = number of 16-wide spatial tiles per workgroup (N = NSP*16).
template <int HIN, int HOUT, int RMAX, int NSP>
__global__ void conv_bn_relu_wmma(const float*  __restrict__ in,     // [B,256,HIN,HIN] f32
                                  const __bf16* __restrict__ apack,  // packed weights
                                  const float*  __restrict__ scale,
                                  const float*  __restrict__ shift,
                                  float* __restrict__ out) {         // [B,256,HOUT,HOUT] f32
  constexpr int WIN  = HIN;
  constexpr int WOUT = HOUT;
  constexpr int M    = HOUT * WOUT;            // spatial outputs per image
  constexpr int SPAN = NSP * 16;               // spatial positions per WG
  constexpr int NT   = (M + SPAN - 1) / SPAN;  // spatial tiles per image

  __shared__ __align__(32) __bf16 patch[RMAX * WIN * 32]; // [row][col][cin32]

  const int blk   = blockIdx.x;
  const int b     = blk / NT;
  const int nt    = blk % NT;
  const int mBase = nt * SPAN;

  const int tid  = threadIdx.x;
  const int wave = tid >> 5;
  const int lane = tid & 31;

  // patch row window covering this spatial span for all 3 kh taps
  const int last = (mBase + SPAN - 1 < M - 1) ? (mBase + SPAN - 1) : (M - 1);
  const int r0   = mBase / WOUT;
  const int R    = (last / WOUT) - r0 + 3;     // <= RMAX

  // per-lane B-fragment coordinates per spatial tile (N = spatial position)
  const int nIdx = lane & 15;
  const int kh16 = lane >> 4;                  // K half (0: K 0-15, 1: K 16-31)
  int ohA[NSP], owA[NSP];
  #pragma unroll
  for (int t = 0; t < NSP; ++t) {
    int pos = mBase + t * 16 + nIdx;
    if (pos > M - 1) pos = M - 1;              // clamp (store is guarded)
    ohA[t] = pos / WOUT;
    owA[t] = pos % WOUT;
  }

  const int j0 = wave * 2;                     // two cout tiles per wave
  const int j1 = wave * 2 + 1;

  v8f acc0[NSP], acc1[NSP];
  #pragma unroll
  for (int t = 0; t < NSP; ++t) { acc0[t] = (v8f)(0.f); acc1[t] = (v8f)(0.f); }

  const float* inb = in + ((size_t)b * NCH) * (HIN * WIN);

  for (int c32 = 0; c32 < 8; ++c32) {
    __syncthreads();
    // ---- stage fp32 -> bf16 patch into LDS: 32 cins x R rows x WIN cols ----
    const float* src = inb + (size_t)(c32 * 32) * (HIN * WIN);
    const int total = R * WIN;
    for (int idx = tid; idx < 32 * total; idx += 256) {
      int cin = idx / total;
      int rc  = idx - cin * total;             // (row-r0)*WIN + col
      int row = rc / WIN + r0;
      int col = rc - (rc / WIN) * WIN;
      float v = src[(size_t)cin * (HIN * WIN) + row * WIN + col];
      patch[rc * 32 + cin] = (__bf16)v;
    }
    if (c32 < 7) __builtin_prefetch(src + 32 * HIN * WIN + tid * 16, 0, 1);
    __syncthreads();

    // ---- 9 taps x (2 cout tiles x NSP spatial tiles) of WMMA ----
    #pragma unroll
    for (int tap = 0; tap < 9; ++tap) {
      const int kh = tap / 3, kw = tap % 3;
      const int q  = tap * 8 + c32;
      const v16bf a0 =
          *(const v16bf*)(apack + ((size_t)((q * 16 + j0) * 32 + lane)) * 16);
      const v16bf a1 =
          *(const v16bf*)(apack + ((size_t)((q * 16 + j1) * 32 + lane)) * 16);
      #pragma unroll
      for (int t = 0; t < NSP; ++t) {
        const int rowLocal = ohA[t] - r0 + kh;
        const int col      = owA[t] + kw;
        const v16bf bfrag =
            *(const v16bf*)&patch[(rowLocal * WIN + col) * 32 + kh16 * 16];
        acc0[t] = __builtin_amdgcn_wmma_f32_16x16x32_bf16(
            false, a0, false, bfrag, (short)0, acc0[t], false, false);
        acc1[t] = __builtin_amdgcn_wmma_f32_16x16x32_bf16(
            false, a1, false, bfrag, (short)0, acc1[t], false, false);
      }
    }
  }

  // ---- BN + ReLU + store. D layout: elem r -> cout = j*16 + (lane>>4)*8 + r,
  //      column (spatial) = tile base + (lane&15). ----
  #pragma unroll
  for (int t = 0; t < NSP; ++t) {
    const int sp = mBase + t * 16 + nIdx;
    if (sp < M) {
      float* ob = out + ((size_t)b * NCH) * M + sp;
      #pragma unroll
      for (int r = 0; r < 8; ++r) {
        int c0 = j0 * 16 + kh16 * 8 + r;
        float v0 = acc0[t][r] * scale[c0] + shift[c0];
        ob[(size_t)c0 * M] = v0 > 0.f ? v0 : 0.f;
        int c1 = j1 * 16 + kh16 * 8 + r;
        float v1 = acc1[t][r] * scale[c1] + shift[c1];
        ob[(size_t)c1 * M] = v1 > 0.f ? v1 : 0.f;
      }
    }
  }
}

// ---------------- Depthwise cross-correlation: 29x29 (x) 5x5 -> 25x25 -----
__global__ void xcorr_kernel(const float* __restrict__ s,   // [B*C, 29,29]
                             const float* __restrict__ k,   // [B*C, 5,5]
                             float* __restrict__ out) {     // [B*C, 25,25]
  const int bc = blockIdx.x;
  const float* sp = s + (size_t)bc * (29 * 29);
  const float* kp = k + (size_t)bc * 25;
  float* op = out + (size_t)bc * (25 * 25);

  __shared__ float ls[29 * 29];
  __shared__ float lk[25];

  for (int i = threadIdx.x; i < 29 * 29; i += 256) ls[i] = sp[i];
  if (threadIdx.x < 25) lk[threadIdx.x] = kp[threadIdx.x];
  __syncthreads();

  for (int o = threadIdx.x; o < 25 * 25; o += 256) {
    int oy = o / 25, ox = o % 25;
    float acc = 0.f;
    #pragma unroll
    for (int ky = 0; ky < 5; ++ky)
      #pragma unroll
      for (int kx = 0; kx < 5; ++kx)
        acc = fmaf(ls[(oy + ky) * 29 + ox + kx], lk[ky * 5 + kx], acc);
    op[o] = acc;
  }
}

// ---------------------------------------------------------------------------
extern "C" void kernel_launch(void* const* d_in, const int* in_sizes, int n_in,
                              void* d_out, int out_size, void* d_ws, size_t ws_size,
                              hipStream_t stream) {
  (void)in_sizes; (void)n_in; (void)out_size; (void)ws_size;

  const float* x_kernel  = (const float*)d_in[0];   // [128,256,7,7]
  const float* x_search  = (const float*)d_in[1];   // [128,256,31,31]
  const float* w_kernel  = (const float*)d_in[2];   // [256,256,3,3]
  const float* bnk_gamma = (const float*)d_in[3];
  const float* bnk_beta  = (const float*)d_in[4];
  const float* bnk_mean  = (const float*)d_in[5];
  const float* bnk_var   = (const float*)d_in[6];
  const float* w_search  = (const float*)d_in[7];
  const float* bns_gamma = (const float*)d_in[8];
  const float* bns_beta  = (const float*)d_in[9];
  const float* bns_mean  = (const float*)d_in[10];
  const float* bns_var   = (const float*)d_in[11];
  float* out = (float*)d_out;                       // [128,256,25,25]

  // Workspace layout (~116 MB total)
  char* ws = (char*)d_ws;
  const size_t APACK_BYTES = (size_t)72 * 16 * 32 * 16 * 2;   // 1,179,648
  __bf16* apackK = (__bf16*)(ws);
  __bf16* apackS = (__bf16*)(ws + APACK_BYTES);
  float*  scaleK = (float*)(ws + 2 * APACK_BYTES);
  float*  shiftK = scaleK + NCH;
  float*  scaleS = shiftK + NCH;
  float*  shiftS = scaleS + NCH;
  float*  k_out  = (float*)(ws + 2 * APACK_BYTES + 4 * NCH * sizeof(float));
  //   k_out: 128*256*25   f32 =   3,276,800 B
  float*  s_out  = k_out + (size_t)NB * NCH * 25;
  //   s_out: 128*256*841  f32 = 110,231,552 B

  // 1) fold BN
  bn_prep_kernel<<<1, 256, 0, stream>>>(bnk_gamma, bnk_beta, bnk_mean, bnk_var,
                                        scaleK, shiftK);
  bn_prep_kernel<<<1, 256, 0, stream>>>(bns_gamma, bns_beta, bns_mean, bns_var,
                                        scaleS, shiftS);

  // 2) pack weights to WMMA-A fragment order (bf16)
  const int PACK_N = 72 * 16 * 32 * 16;
  pack_weights_kernel<<<(PACK_N + 255) / 256, 256, 0, stream>>>(w_kernel, apackK);
  pack_weights_kernel<<<(PACK_N + 255) / 256, 256, 0, stream>>>(w_search, apackS);

  // 3) convs (implicit GEMM, WMMA bf16)
  //    kernel branch: 7x7 -> 5x5, M=25, NSP=2 (one WG covers the image), RMAX=7
  conv_bn_relu_wmma<7, 5, 7, 2><<<NB * 1, 256, 0, stream>>>(
      x_kernel, apackK, scaleK, shiftK, k_out);
  //    search branch: 31x31 -> 29x29, M=841, NSP=4 -> 14 spatial tiles, RMAX=6
  conv_bn_relu_wmma<31, 29, 6, 4><<<NB * 14, 256, 0, stream>>>(
      x_search, apackS, scaleS, shiftS, s_out);

  // 4) depthwise cross-correlation
  xcorr_kernel<<<NB * NCH, 256, 0, stream>>>(s_out, k_out, out);
}
